// SSIMLoss_82437602279479
// MI455X (gfx1250) — matrix operand order repaired and checked by reading, not compiled, not run on hardware
//
#include <hip/hip_runtime.h>

typedef __attribute__((ext_vector_type(16))) _Float16 v16h;
typedef __attribute__((ext_vector_type(8)))  _Float16 v8h;
typedef __attribute__((ext_vector_type(8)))  float    v8f;
typedef __attribute__((ext_vector_type(2)))  float    v2f;

#define IMG_H 512
#define IMG_W 512
#define OUT_H 502
#define OUT_W 502
#define NCHANNELS 48            // 16 images * 3 channels
#define TILES_X 32
#define TILES_Y 32
#define WAVES_PER_BLOCK 8
#define C1_CONST 6.5025f        // (0.01*255)^2
#define C2_CONST 58.5225f       // (0.03*255)^2
#define INV_COUNT (1.0 / 12096192.0)     // 48 * 502 * 502
#define BAND_OFFSET 256         // byte offset of packed band table in d_ws

// ds_swizzle group-of-32 xor patterns: offset = (xor<<10) | 0x1f
#define SWZ_X16 0x401F
#define SWZ_X8  0x201F
#define SWZ_X4  0x101F
#define SWZ_X2  0x081F
#define SWZ_X1  0x041F

// 11-tap gaussian, sigma=1.5, normalized: g(d)=exp(-(d-5)^2/4.5)/3.7592286
__device__ __forceinline__ float gval(int d) {
    const float e = (float)((d - 5) * (d - 5)) * -0.2222222222f;
    const float v = __expf(e) * 0.26601204f;
    return (d >= 0 && d <= 10) ? v : 0.0f;
}

union V16U { unsigned int u[8]; v16h h; };
union V8U  { unsigned int u[4]; v8h  h; };

__device__ __forceinline__ float swz_add(float s, int pat_is_x16_to_x1_unused) { return s; }

// Convert four float4s (16 contiguous lane values) into a v16h fragment.
__device__ __forceinline__ v16h to16(float4 a, float4 b, float4 c, float4 d) {
    v16h r{};
    r[0]  = (_Float16)a.x; r[1]  = (_Float16)a.y; r[2]  = (_Float16)a.z; r[3]  = (_Float16)a.w;
    r[4]  = (_Float16)b.x; r[5]  = (_Float16)b.y; r[6]  = (_Float16)b.z; r[7]  = (_Float16)b.w;
    r[8]  = (_Float16)c.x; r[9]  = (_Float16)c.y; r[10] = (_Float16)c.z; r[11] = (_Float16)c.w;
    r[12] = (_Float16)d.x; r[13] = (_Float16)d.y; r[14] = (_Float16)d.z; r[15] = (_Float16)d.w;
    return r;
}

// One-wave setup: zero the f64 accumulator and pre-pack the two Toeplitz band
// fragments (pure function of lane id) so the hot kernel just reloads them.
__global__ void ssim_setup_kernel(double* acc, v16h* bands) {
    const int lane = threadIdx.x & 31;
    if (threadIdx.x == 0) acc[0] = 0.0;
    const int r  = lane & 15;
    const int hi = (lane >> 4) & 1;
    v16h bandA{};   // pass-2 A: g[j - r], A layout: K = h+16*hi
    v16h bandB{};   // pass-1 B: g[j - c], B layout: K = (h<8?h:h+8)+8*hi
#pragma unroll
    for (int h = 0; h < 16; ++h) {
        const int jA = h + 16 * hi;
        const int jB = (h < 8 ? h : h + 8) + 8 * hi;
        bandA[h] = (_Float16)gval(jA - r);
        bandB[h] = (_Float16)gval(jB - r);
    }
    bands[lane * 2 + 0] = bandA;
    bands[lane * 2 + 1] = bandB;
}

__global__ void ssim_finish_kernel(const double* acc, float* out) {
    out[0] = (float)(1.0 - acc[0] * INV_COUNT);
}

__global__ __launch_bounds__(32 * WAVES_PER_BLOCK)
void ssim_tile_kernel(const float* __restrict__ x, const float* __restrict__ y,
                      double* __restrict__ acc, const v16h* __restrict__ bands) {
    const int lane = threadIdx.x & 31;
    const int wv   = threadIdx.x >> 5;
    const int tile = blockIdx.x * WAVES_PER_BLOCK + wv;
    const int tx   = tile & (TILES_X - 1);
    const int ty   = (tile / TILES_X) & (TILES_Y - 1);
    const int ch   = tile / (TILES_X * TILES_Y);

    const float* px = x + (size_t)ch * (IMG_H * IMG_W);
    const float* py = y + (size_t)ch * (IMG_H * IMG_W);
    const int gy0 = ty * 16;
    const int gx0 = tx * 16;

    const int r16 = lane & 15;
    const int hi  = (lane >> 4) & 1;

    // L2-hot reload of the pre-packed constant band fragments (2x b128 each).
    const v16h bandA = bands[lane * 2 + 0];
    const v16h bandB = bands[lane * 2 + 1];

    // Column bases for this lane's two contiguous 8-float K-runs.
    // Clamped loads stay in-bounds; distorted values only reach zero-tap K
    // positions or outputs masked at the reduction.
    const int cb0 = min(gx0 + 8 * hi,      IMG_W - 8);
    const int cb1 = min(gx0 + 8 * hi + 16, IMG_W - 8);

    // ---- build all 10 pass-1 fragments first (gives the scheduler
    // independent work to hide WMMA D->A/B hazards).
    v16h fr[5][2];
#pragma unroll
    for (int t = 0; t < 2; ++t) {
        const int row = min(gy0 + r16 + 16 * t, IMG_H - 1);
        const float* rowX = px + (size_t)row * IMG_W;
        const float* rowY = py + (size_t)row * IMG_W;

        const float4 xa = *(const float4*)(rowX + cb0);
        const float4 xb = *(const float4*)(rowX + cb0 + 4);
        const float4 xc = *(const float4*)(rowX + cb1);
        const float4 xd = *(const float4*)(rowX + cb1 + 4);
        const float4 ya = *(const float4*)(rowY + cb0);
        const float4 yb = *(const float4*)(rowY + cb0 + 4);
        const float4 yc = *(const float4*)(rowY + cb1);
        const float4 yd = *(const float4*)(rowY + cb1 + 4);

        const v16h ax = to16(xa, xb, xc, xd);
        const v16h ay = to16(ya, yb, yc, yd);
        fr[0][t] = ax;
        fr[1][t] = ay;
        fr[2][t] = ax * ax;        // v_pk_mul_f16
        fr[3][t] = ay * ay;
        fr[4][t] = ax * ay;
    }

    // ---- pass 1: HORIZONTAL blur, f16 accumulator.
    // M[m][c] = sum_j I[gy0+m][gx0+j] * g[j-c];  A = image rows (row-major),
    // B = constant band.  f16 D layout: lane = c, dword v = rows m = 2v,2v+1
    // (+8 for hi lanes) -> dwords are already the packed pairs pass-2 B needs.
    const v8h czero{};
    v8h vt16[5][2];
#pragma unroll
    for (int q = 0; q < 5; ++q) {
#pragma unroll
        for (int t = 0; t < 2; ++t) {
            vt16[q][t] = __builtin_amdgcn_wmma_f16_16x16x32_f16(
                false, fr[q][t], false, bandB, (short)0, czero, false, false);
        }
    }

    // ---- pass 2: VERTICAL blur: O = Band x M (f32 accumulator).
    // Repack 16-bit D dwords into B dwords: one immediate ds_swizzle (xor-16
    // half swap, no address register) + select each.
    const v8f zero{};
    v8f outq[5];
#pragma unroll
    for (int q = 0; q < 5; ++q) {
        V8U d0, d1;
        d0.h = vt16[q][0];
        d1.h = vt16[q][1];
        V16U ub;
#pragma unroll
        for (int d = 0; d < 4; ++d) {
            const unsigned int s0 = (unsigned int)__builtin_amdgcn_ds_swizzle((int)d0.u[d], SWZ_X16);
            const unsigned int s1 = (unsigned int)__builtin_amdgcn_ds_swizzle((int)d1.u[d], SWZ_X16);
            ub.u[d]     = hi ? s1 : d0.u[d];   // j = 0..7   | 16..23
            ub.u[d + 4] = hi ? d1.u[d] : s0;   // j = 8..15  | 24..31
        }
        outq[q] = __builtin_amdgcn_wmma_f32_16x16x32_f16(
            false, bandA, false, ub.h, (short)0, zero, false, false);
    }

    // ---- SSIM map + masked sum, packed 2-wide so mul/add land on v_pk_*_f32.
    // D element: (row r = v + 8*hi, col c = lane%16)
    const int ox = gx0 + r16;
    const bool colok = (ox < OUT_W);
    v2f ssv = {0.0f, 0.0f};
#pragma unroll
    for (int p = 0; p < 4; ++p) {
        const int e0 = 2 * p, e1 = 2 * p + 1;
        const v2f mu1 = {outq[0][e0], outq[0][e1]};
        const v2f mu2 = {outq[1][e0], outq[1][e1]};
        const v2f xx  = {outq[2][e0], outq[2][e1]};
        const v2f yy  = {outq[3][e0], outq[3][e1]};
        const v2f xy  = {outq[4][e0], outq[4][e1]};
        const v2f mu1sq = mu1 * mu1;
        const v2f mu2sq = mu2 * mu2;
        const v2f mu12  = mu1 * mu2;
        const v2f num = (2.0f * mu12 + C1_CONST) * (2.0f * (xy - mu12) + C2_CONST);
        const v2f den = (mu1sq + mu2sq + C1_CONST) *
                        ((xx - mu1sq) + (yy - mu2sq) + C2_CONST);
        const float ss0 = num[0] * __builtin_amdgcn_rcpf(den[0]);
        const float ss1 = num[1] * __builtin_amdgcn_rcpf(den[1]);
        const int oy0 = gy0 + e0 + 8 * hi;
        const int oy1 = oy0 + 1;
        const v2f add = {(colok && oy0 < OUT_H) ? ss0 : 0.0f,
                         (colok && oy1 < OUT_H) ? ss1 : 0.0f};
        ssv += add;
    }
    float ssum = ssv[0] + ssv[1];

    // wave32 tree reduction via immediate ds_swizzle + one f64 atomic per wave
    ssum += __int_as_float(__builtin_amdgcn_ds_swizzle(__float_as_int(ssum), SWZ_X16));
    ssum += __int_as_float(__builtin_amdgcn_ds_swizzle(__float_as_int(ssum), SWZ_X8));
    ssum += __int_as_float(__builtin_amdgcn_ds_swizzle(__float_as_int(ssum), SWZ_X4));
    ssum += __int_as_float(__builtin_amdgcn_ds_swizzle(__float_as_int(ssum), SWZ_X2));
    ssum += __int_as_float(__builtin_amdgcn_ds_swizzle(__float_as_int(ssum), SWZ_X1));
    if (lane == 0) atomicAdd(acc, (double)ssum);
}

extern "C" void kernel_launch(void* const* d_in, const int* in_sizes, int n_in,
                              void* d_out, int out_size, void* d_ws, size_t ws_size,
                              hipStream_t stream) {
    const float* x = (const float*)d_in[0];
    const float* y = (const float*)d_in[1];
    float* out = (float*)d_out;
    double* acc = (double*)d_ws;
    v16h* bands = (v16h*)((char*)d_ws + BAND_OFFSET);

    const int total_tiles = NCHANNELS * TILES_X * TILES_Y;      // 49152
    const int blocks = total_tiles / WAVES_PER_BLOCK;           // 6144

    ssim_setup_kernel<<<1, 32, 0, stream>>>(acc, bands);
    ssim_tile_kernel<<<blocks, 32 * WAVES_PER_BLOCK, 0, stream>>>(x, y, acc, bands);
    ssim_finish_kernel<<<1, 1, 0, stream>>>(acc, out);
}
